// BipolarSpikeSelfAttention_68401649156493
// MI455X (gfx1250) — matrix operand order repaired and checked by reading, not compiled, not run on hardware
//
#include <hip/hip_runtime.h>
#include <hip/hip_bf16.h>

// ---------------------------------------------------------------------------
// BipolarSpikeSelfAttention on MI455X (gfx1250), wave32 + WMMA f16->f32.
// Pipeline: f32->f16 convert, 3x WMMA GEMM (Q/K/V proj, bipolar+scale folded
// into epilogue), flash-attention WMMA kernel (async-to-LDS K staging when
// available), WMMA GEMM output projection.
// ---------------------------------------------------------------------------

typedef __attribute__((ext_vector_type(16))) _Float16 v16h;
typedef __attribute__((ext_vector_type(8)))  _Float16 v8h;
typedef __attribute__((ext_vector_type(8)))  float    v8f;

#define D_MODEL  1024
#define NHEAD    16
#define HEAD_DIM 64
#define BATCH    2
#define SEQ      2048
#define MTOT     (BATCH * SEQ)   // 4096 rows total

#if __has_builtin(__builtin_amdgcn_global_load_async_to_lds_b128) && \
    __has_builtin(__builtin_amdgcn_s_wait_asynccnt)
#define USE_ASYNC_LDS 1
// Builtin prototype (from compiler diagnostic): pointers to 16B int vectors,
// src in global (AS1), dst in LDS (AS3).
typedef int v4i_vs __attribute__((vector_size(16)));
typedef __attribute__((address_space(1))) v4i_vs glob_v4i_t;
typedef __attribute__((address_space(3))) v4i_vs lds_v4i_t;
#else
#define USE_ASYNC_LDS 0
#endif

// ------------------------- WMMA helpers ------------------------------------

__device__ __forceinline__ v8f wmma_f32(v16h a, v16h b, v8f c) {
  // D = A(16x32 f16) * B(32x16 f16) + C(16x16 f32)
  return __builtin_amdgcn_wmma_f32_16x16x32_f16(
      /*neg_a=*/false, a, /*neg_b=*/false, b,
      /*c_mod=*/(short)0, c, /*reuse_a=*/false, /*reuse_b=*/false);
}

// A-matrix fragment (16x32, f16), source row-major with leading dim `ld`.
// Layout (ISA 7.12.2): lane L holds row m=L&15; element i -> k =
// kBase + 16*(i>>3) + (i&7) + 8*(L>>4). Elements 0..7 and 8..15 are each
// 8 contiguous halves -> two 16B loads.
__device__ __forceinline__ v16h load_a_frag(const _Float16* __restrict__ base,
                                            int ld, int kBase, int lane) {
  const int m = lane & 15;
  const int hi = lane >> 4;
  const _Float16* p = base + (size_t)m * ld + kBase + hi * 8;
  v8h lo = *reinterpret_cast<const v8h*>(p);
  v8h hh = *reinterpret_cast<const v8h*>(p + 16);
  v16h a;
#pragma unroll
  for (int i = 0; i < 8; ++i) { a[i] = lo[i]; a[i + 8] = hh[i]; }
  return a;
}

// B-matrix fragment (32x16, f16). Source is row-major where row n supplies
// B column n (i.e. W[n][k] for GEMM with B = W^T, or K[key][dim] for QK^T).
// Layout: lane L holds column n=L&15; element i -> k =
// kBase + 16*(i>>3) + (i&7) + 8*(L>>4) -> two 16B loads.
__device__ __forceinline__ v16h load_b_frag(const _Float16* __restrict__ base,
                                            int ld, int kBase, int lane) {
  const int n = lane & 15;
  const int hi = lane >> 4;
  const _Float16* p = base + (size_t)n * ld + kBase + hi * 8;
  v8h lo = *reinterpret_cast<const v8h*>(p);
  v8h hh = *reinterpret_cast<const v8h*>(p + 16);
  v16h b;
#pragma unroll
  for (int i = 0; i < 8; ++i) { b[i] = lo[i]; b[i + 8] = hh[i]; }
  return b;
}

__device__ __forceinline__ v8f vzero8() {
  v8f z = {0.f, 0.f, 0.f, 0.f, 0.f, 0.f, 0.f, 0.f};
  return z;
}

// ------------------------- f32 -> f16 convert -------------------------------

__global__ __launch_bounds__(256) void cvt8_f32_f16(
    const float* __restrict__ in, _Float16* __restrict__ out, int n8) {
  int i = blockIdx.x * blockDim.x + threadIdx.x;
  if (i >= n8) return;
  const float* p = in + (size_t)i * 8;
  v8h o;
#pragma unroll
  for (int j = 0; j < 8; ++j) o[j] = (_Float16)p[j];
  *reinterpret_cast<v8h*>(out + (size_t)i * 8) = o;
}

// ------------------------- GEMM: out = alpha*(A @ W^T + bias) + beta --------
// A: [M,K] f16 row-major; W: [N,K] f16 row-major (so B = W^T);
// out: f16 [M,N] (outF32==0) or f32 [M,N] (outF32==1).
// Block: 128 threads = 4 waves (2x2); each wave -> 64x64 output (4x4 WMMA
// tiles, 16 f32 accumulators); block tile 128x128. Per 32-k step a wave does
// 16 b128 loads and 16 WMMAs (1:1), vs 2:1 for a 32x32 wave tile.

__global__ __launch_bounds__(128) void gemm_wmma(
    const _Float16* __restrict__ A, const _Float16* __restrict__ W,
    const float* __restrict__ bias, void* __restrict__ outPtr,
    int M, int K, int N, float alpha, float beta, int outF32) {
  const int lane = threadIdx.x & 31;
  const int wave = threadIdx.x >> 5;      // 0..3
  const int wr = wave >> 1;               // 0..1
  const int wc = wave & 1;                // 0..1
  const int rowBase = blockIdx.x * 128 + wr * 64;
  const int colBase = blockIdx.y * 128 + wc * 64;

  v8f acc[4][4];
#pragma unroll
  for (int i = 0; i < 4; ++i)
#pragma unroll
    for (int j = 0; j < 4; ++j) acc[i][j] = vzero8();

  const _Float16* Ar[4];
  const _Float16* Wc[4];
#pragma unroll
  for (int i = 0; i < 4; ++i) {
    Ar[i] = A + (size_t)(rowBase + 16 * i) * K;
    Wc[i] = W + (size_t)(colBase + 16 * i) * K;
  }

  for (int kB = 0; kB < K; kB += 32) {
    v16h af[4], bf[4];
#pragma unroll
    for (int i = 0; i < 4; ++i) af[i] = load_a_frag(Ar[i], K, kB, lane);
#pragma unroll
    for (int j = 0; j < 4; ++j) bf[j] = load_b_frag(Wc[j], K, kB, lane);
#pragma unroll
    for (int i = 0; i < 4; ++i)
#pragma unroll
      for (int j = 0; j < 4; ++j) acc[i][j] = wmma_f32(af[i], bf[j], acc[i][j]);
  }

  const int hi = lane >> 4;
  const int n0 = lane & 15;
  // Epilogue: y = alpha*(acc + bias[col]) + beta
#pragma unroll
  for (int tr = 0; tr < 4; ++tr) {
#pragma unroll
    for (int tc = 0; tc < 4; ++tc) {
      const int col = colBase + tc * 16 + n0;
      const float bv = bias[col];
#pragma unroll
      for (int j = 0; j < 8; ++j) {
        const int row = rowBase + tr * 16 + hi * 8 + j;
        const float y = alpha * (acc[tr][tc][j] + bv) + beta;
        if (outF32) {
          reinterpret_cast<float*>(outPtr)[(size_t)row * N + col] = y;
        } else {
          reinterpret_cast<_Float16*>(outPtr)[(size_t)row * N + col] =
              (_Float16)y;
        }
      }
    }
  }
}

// ------------------------- Flash attention ----------------------------------
// Q already holds 0.25*q - 0.125  (bipolar * 1/sqrt(Dh) folded in)
// K already holds 2*k - 1         (bipolar)
// Grid: (SEQ/64, NHEAD, BATCH); block 128 threads = 4 waves.
// Each wave owns 16 query rows; block iterates key tiles of 32 together.

__global__ __launch_bounds__(128) void attn_wmma(
    const _Float16* __restrict__ Q, const _Float16* __restrict__ Km,
    const _Float16* __restrict__ V, _Float16* __restrict__ O) {
  __shared__ __align__(16) _Float16 Ks[32][72];       // [key][dim], padded
  __shared__ __align__(16) _Float16 Vt[64][40];       // [dim][key], padded
  __shared__ __align__(16) _Float16 Ps[4][16][32];    // per-wave P scratch

  const int qBlk = blockIdx.x;   // 64-row query block
  const int h = blockIdx.y;
  const int b = blockIdx.z;
  const int tid = threadIdx.x;
  const int lane = tid & 31;
  const int wave = tid >> 5;     // 0..3
  const int hi = lane >> 4;
  const int ln = lane & 15;

  const size_t rowQ0 = (size_t)b * SEQ + qBlk * 64 + wave * 16;
  const _Float16* Qbase = Q + rowQ0 * D_MODEL + h * HEAD_DIM;

  // Query A-fragments for this wave's 16 rows: K-dim = 64 -> 2 fragments.
  v16h aQ0 = load_a_frag(Qbase, D_MODEL, 0, lane);
  v16h aQ1 = load_a_frag(Qbase, D_MODEL, 32, lane);

  v8f o0 = vzero8(), o1 = vzero8(), o2 = vzero8(), o3 = vzero8();
  float mrow[8], lrow[8];
#pragma unroll
  for (int j = 0; j < 8; ++j) { mrow[j] = -1e30f; lrow[j] = 0.f; }

  const int ldKey = tid >> 2;          // 0..31
  const int ldDim = (tid & 3) * 16;    // 0,16,32,48

  for (int kt = 0; kt < SEQ / 32; ++kt) {
    // --- cooperative stage of K tile (as-is) and V tile (transposed) ---
    {
      const size_t keyRow = (size_t)b * SEQ + kt * 32 + ldKey;
      const _Float16* kp = Km + keyRow * D_MODEL + h * HEAD_DIM + ldDim;
#if USE_ASYNC_LDS
      // CDNA5 async copy: global -> LDS without a VGPR round trip
      // (tracked by ASYNCcnt). 32 bytes per thread = 2x b128.
      glob_v4i_t* gk = (glob_v4i_t*)(unsigned long long)kp;
      lds_v4i_t* lk =
          (lds_v4i_t*)(unsigned)(unsigned long long)&Ks[ldKey][ldDim];
      __builtin_amdgcn_global_load_async_to_lds_b128(gk, lk, 0, 0);
      __builtin_amdgcn_global_load_async_to_lds_b128(gk + 1, lk + 1, 0, 0);
#else
      v8h k0 = *reinterpret_cast<const v8h*>(kp);
      v8h k1 = *reinterpret_cast<const v8h*>(kp + 8);
      *reinterpret_cast<v8h*>(&Ks[ldKey][ldDim]) = k0;
      *reinterpret_cast<v8h*>(&Ks[ldKey][ldDim + 8]) = k1;
#endif
      const _Float16* vp = V + keyRow * D_MODEL + h * HEAD_DIM + ldDim;
      v8h v0 = *reinterpret_cast<const v8h*>(vp);
      v8h v1 = *reinterpret_cast<const v8h*>(vp + 8);
#pragma unroll
      for (int i = 0; i < 8; ++i) Vt[ldDim + i][ldKey] = v0[i];
#pragma unroll
      for (int i = 0; i < 8; ++i) Vt[ldDim + 8 + i][ldKey] = v1[i];
    }
#if USE_ASYNC_LDS
    __builtin_amdgcn_s_wait_asynccnt(0);
#endif
    __syncthreads();

    // --- S tile: 16 q-rows x 32 keys (two 16x16 WMMA D tiles) ---
    v8f s0 = vzero8(), s1 = vzero8();
    {
      v16h bk;
      bk = load_b_frag(&Ks[0][0], 72, 0, lane);   s0 = wmma_f32(aQ0, bk, s0);
      bk = load_b_frag(&Ks[0][0], 72, 32, lane);  s0 = wmma_f32(aQ1, bk, s0);
      bk = load_b_frag(&Ks[16][0], 72, 0, lane);  s1 = wmma_f32(aQ0, bk, s1);
      bk = load_b_frag(&Ks[16][0], 72, 32, lane); s1 = wmma_f32(aQ1, bk, s1);
    }

    // --- online softmax: rows live in element j (row = j + 8*hi),
    //     columns across the 16 lanes of this half-wave ---
    float corr[8];
#pragma unroll
    for (int j = 0; j < 8; ++j) {
      float mx = fmaxf(s0[j], s1[j]);
#pragma unroll
      for (int off = 1; off < 16; off <<= 1)
        mx = fmaxf(mx, __shfl_xor(mx, off, 32));
      const float mnew = fmaxf(mrow[j], mx);
      corr[j] = __expf(mrow[j] - mnew);
      mrow[j] = mnew;
      const float p0 = __expf(s0[j] - mnew);
      const float p1 = __expf(s1[j] - mnew);
      float sum = p0 + p1;
#pragma unroll
      for (int off = 1; off < 16; off <<= 1)
        sum += __shfl_xor(sum, off, 32);
      lrow[j] = lrow[j] * corr[j] + sum;
      // Spill P (D-layout) to per-wave LDS scratch for A-layout reload.
      Ps[wave][hi * 8 + j][ln] = (_Float16)p0;
      Ps[wave][hi * 8 + j][ln + 16] = (_Float16)p1;
    }
    // rescale running output
#pragma unroll
    for (int j = 0; j < 8; ++j) {
      o0[j] *= corr[j]; o1[j] *= corr[j]; o2[j] *= corr[j]; o3[j] *= corr[j];
    }

    // --- O += P(16x32) @ V(32x64): 4 output dim-tiles ---
    // Same-wave LDS write->read is in-order, no barrier needed here.
    v16h aP = load_a_frag(&Ps[wave][0][0], 32, 0, lane);
    {
      v16h bv;
      bv = load_b_frag(&Vt[0][0], 40, 0, lane);  o0 = wmma_f32(aP, bv, o0);
      bv = load_b_frag(&Vt[16][0], 40, 0, lane); o1 = wmma_f32(aP, bv, o1);
      bv = load_b_frag(&Vt[32][0], 40, 0, lane); o2 = wmma_f32(aP, bv, o2);
      bv = load_b_frag(&Vt[48][0], 40, 0, lane); o3 = wmma_f32(aP, bv, o3);
    }
    __syncthreads();  // protect Ks/Vt before next tile overwrites
  }

  // --- finalize: divide by l and store (f16, merged-head layout) ---
  _Float16* Obase = O + rowQ0 * D_MODEL + h * HEAD_DIM;
#pragma unroll
  for (int j = 0; j < 8; ++j) {
    const float inv = 1.0f / lrow[j];
    const size_t r = (size_t)(hi * 8 + j) * D_MODEL;
    Obase[r + 0 + ln]  = (_Float16)(o0[j] * inv);
    Obase[r + 16 + ln] = (_Float16)(o1[j] * inv);
    Obase[r + 32 + ln] = (_Float16)(o2[j] * inv);
    Obase[r + 48 + ln] = (_Float16)(o3[j] * inv);
  }
}

// ------------------------- host-side launch ---------------------------------

extern "C" void kernel_launch(void* const* d_in, const int* in_sizes, int n_in,
                              void* d_out, int out_size, void* d_ws,
                              size_t ws_size, hipStream_t stream) {
  const float* x  = (const float*)d_in[0];
  const float* Wq = (const float*)d_in[1];
  const float* bq = (const float*)d_in[2];
  const float* Wk = (const float*)d_in[3];
  const float* bk = (const float*)d_in[4];
  const float* Wv = (const float*)d_in[5];
  const float* bv = (const float*)d_in[6];
  const float* Wo = (const float*)d_in[7];
  const float* bo = (const float*)d_in[8];

  const size_t XE = (size_t)MTOT * D_MODEL;     // 4M halves
  const size_t WE = (size_t)D_MODEL * D_MODEL;  // 1M halves

  _Float16* ws  = (_Float16*)d_ws;
  _Float16* xh  = ws;             // [MTOT, D] x in f16
  _Float16* wqh = xh + XE;
  _Float16* wkh = wqh + WE;
  _Float16* wvh = wkh + WE;
  _Float16* woh = wvh + WE;
  _Float16* qb  = woh + WE;       // scaled-bipolar Q
  _Float16* kb  = qb + XE;        // bipolar K
  _Float16* vh  = kb + XE;        // V
  _Float16* ah  = vh + XE;        // attention output (pre Wo)

  // 1) convert inputs to f16
  cvt8_f32_f16<<<dim3((unsigned)((XE / 8 + 255) / 256)), 256, 0, stream>>>(
      x, xh, (int)(XE / 8));
  cvt8_f32_f16<<<dim3((unsigned)((WE / 8 + 255) / 256)), 256, 0, stream>>>(
      Wq, wqh, (int)(WE / 8));
  cvt8_f32_f16<<<dim3((unsigned)((WE / 8 + 255) / 256)), 256, 0, stream>>>(
      Wk, wkh, (int)(WE / 8));
  cvt8_f32_f16<<<dim3((unsigned)((WE / 8 + 255) / 256)), 256, 0, stream>>>(
      Wv, wvh, (int)(WE / 8));
  cvt8_f32_f16<<<dim3((unsigned)((WE / 8 + 255) / 256)), 256, 0, stream>>>(
      Wo, woh, (int)(WE / 8));

  // 2) projections (bipolar + 1/sqrt(Dh) folded into Q/K epilogues)
  dim3 gb(MTOT / 128, D_MODEL / 128);  // 32 x 8, 128x128 block tiles
  gemm_wmma<<<gb, 128, 0, stream>>>(xh, wqh, bq, qb, MTOT, D_MODEL, D_MODEL,
                                    0.25f, -0.125f, 0);   // (2q-1)/8
  gemm_wmma<<<gb, 128, 0, stream>>>(xh, wkh, bk, kb, MTOT, D_MODEL, D_MODEL,
                                    2.0f, -1.0f, 0);      // 2k-1
  gemm_wmma<<<gb, 128, 0, stream>>>(xh, wvh, bv, vh, MTOT, D_MODEL, D_MODEL,
                                    1.0f, 0.0f, 0);

  // 3) flash attention per (batch, head, 64-row query block)
  attn_wmma<<<dim3(SEQ / 64, NHEAD, BATCH), 128, 0, stream>>>(qb, kb, vh, ah);

  // 4) output projection -> f32 d_out
  gemm_wmma<<<gb, 128, 0, stream>>>(ah, woh, bo, d_out, MTOT, D_MODEL, D_MODEL,
                                    1.0f, 0.0f, 1);
}
